// GPT2Attention_29231547416971
// MI455X (gfx1250) — compile-verified, matrix operand order checked by reading, not run on hardware
//
#include <hip/hip_runtime.h>

#define HIDDEN 1024
#define NHEAD  16
#define HSIZE  64
#define BATCH  2
#define SEQ    2048
#define TOK    (BATCH * SEQ)   // 4096 rows
#define KDIM   1024            // reduction dim for both GEMMs
#define LDK    40              // padded LDS row (32 bf16 + 8 pad) -> 80B, 16B aligned

typedef __attribute__((ext_vector_type(16))) __bf16 v16bf;
typedef __attribute__((ext_vector_type(8)))  float  v8f;

struct alignas(16) B128 { unsigned long long lo, hi; };

// A/B fragment for v_wmma_f32_16x16x32_bf16 from row-major [16 x K] region.
// lane&15 -> row (M for A, N for B); kg=(lane>=16)*8; per lane K = kg..kg+7
// (VGPR0-3) and 16+kg..23+kg (VGPR4-7)  => two contiguous 16B chunks.
__device__ __forceinline__ v16bf load_frag(const __bf16* p, int stride, int lane) {
    const int r  = lane & 15;
    const int kg = (lane >> 4) << 3;
    const __bf16* q = p + r * stride + kg;
    B128 c0 = *(const B128*)(q);
    B128 c1 = *(const B128*)(q + 16);
    v16bf f;
    __builtin_memcpy(&f, &c0, 16);
    __builtin_memcpy((char*)&f + 16, &c1, 16);
    return f;
}

__device__ __forceinline__ v8f wmma_bf16(v16bf a, v16bf b, v8f c) {
    return __builtin_amdgcn_wmma_f32_16x16x32_bf16(false, a, false, b, (short)0, c,
                                                   false, false);
}

// Async-copy one 128x32 bf16 tile of A and of B into LDS via the CDNA5
// ASYNCcnt path (no VGPR staging). 512 16B-chunks per tile, 256 threads ->
// 2 chunks/thread/tile, fully unrolled. LDS offset = low 32 bits of the flat
// shared address (flat->LDS mapping truncates addr[31:0]).
__device__ __forceinline__ void async_fill_tile(const __bf16* __restrict__ A,
                                                const __bf16* __restrict__ B,
                                                __bf16* lA, __bf16* lB,
                                                int mBase, int nBase, int k0, int tid) {
#pragma unroll
    for (int t = 0; t < 2; ++t) {
        int ci  = tid + t * 256;
        int row = ci >> 2;
        int kc  = (ci & 3) << 3;
        const __bf16* ga = A + (size_t)(mBase + row) * KDIM + k0 + kc;
        const __bf16* gb = B + (size_t)(nBase + row) * KDIM + k0 + kc;
        unsigned la = (unsigned)(size_t)(lA + row * LDK + kc);
        unsigned lb = (unsigned)(size_t)(lB + row * LDK + kc);
        asm volatile("global_load_async_to_lds_b128 %0, %1, off"
                     :: "v"(la), "v"(ga) : "memory");
        asm volatile("global_load_async_to_lds_b128 %0, %1, off"
                     :: "v"(lb), "v"(gb) : "memory");
    }
}

// ---------------------------------------------------------------------------
// f32 -> bf16 conversion (4 elements / thread)
// ---------------------------------------------------------------------------
__global__ __launch_bounds__(256) void cvt_f32_to_bf16(const float* __restrict__ src,
                                                       __bf16* __restrict__ dst, int n) {
    int i = (blockIdx.x * 256 + threadIdx.x) * 4;
    if (i + 3 < n) {
        const float4 v = *(const float4*)(src + i);
        dst[i + 0] = (__bf16)v.x;
        dst[i + 1] = (__bf16)v.y;
        dst[i + 2] = (__bf16)v.z;
        dst[i + 3] = (__bf16)v.w;
    }
}

// ---------------------------------------------------------------------------
// C[M,N] = A[M,K] * B[N,K]^T + bias[N]  (x @ W^T, both row-major bf16).
// 128x128 block, 256 threads = 8 waves (4 along M x 2 along N), wave tile
// 32x64 = 2x4 WMMA accum tiles. Double-buffered LDS with async-to-LDS copies
// prefetching K-tile t+1 while WMMAs consume tile t.
// mode 0: scatter into Q[b,h,s,c], K[b,h,s,c], Vt[b,h,c,s]  (N = 3*HIDDEN)
// mode 1: f32 row-major output [M, HIDDEN]                  (N = HIDDEN)
// ---------------------------------------------------------------------------
__global__ __launch_bounds__(256) void gemm_bf16_wmma(
    const __bf16* __restrict__ A, const __bf16* __restrict__ B,
    const float* __restrict__ bias, int N, int mode,
    __bf16* __restrict__ Qh, __bf16* __restrict__ Kh, __bf16* __restrict__ Vt,
    float* __restrict__ Out) {
    __shared__ __bf16 ldsbuf[2 * 2 * 128 * LDK];  // [buf][A|B][128*LDK] = 40 KB

    const int tid   = threadIdx.x;
    const int lane  = tid & 31;
    const int wid   = tid >> 5;
    const int waveM = wid & 3;   // 4 waves along M (32 rows each)
    const int waveN = wid >> 2;  // 2 waves along N (64 cols each)
    const int mBase = blockIdx.y * 128;
    const int nBase = blockIdx.x * 128;

    v8f acc[2][4] = {};

    // prologue: stage K-tile 0 into buffer 0
    async_fill_tile(A, B, ldsbuf, ldsbuf + 128 * LDK, mBase, nBase, 0, tid);
    asm volatile("s_wait_asynccnt 0" ::: "memory");
    __syncthreads();

    int cur = 0;
    for (int k0 = 0; k0 < KDIM; k0 += 32) {
        __bf16* curA = ldsbuf + cur * (2 * 128 * LDK);
        __bf16* curB = curA + 128 * LDK;
        if (k0 + 32 < KDIM) {  // async prefetch of next tile overlaps compute
            __bf16* nxtA = ldsbuf + (cur ^ 1) * (2 * 128 * LDK);
            async_fill_tile(A, B, nxtA, nxtA + 128 * LDK, mBase, nBase, k0 + 32, tid);
        }

        v16bf afrag[2], bfrag[4];
#pragma unroll
        for (int i = 0; i < 2; ++i)
            afrag[i] = load_frag(&curA[(waveM * 32 + i * 16) * LDK], LDK, lane);
#pragma unroll
        for (int j = 0; j < 4; ++j)
            bfrag[j] = load_frag(&curB[(waveN * 64 + j * 16) * LDK], LDK, lane);
#pragma unroll
        for (int i = 0; i < 2; ++i)
#pragma unroll
            for (int j = 0; j < 4; ++j)
                acc[i][j] = wmma_bf16(afrag[i], bfrag[j], acc[i][j]);

        asm volatile("s_wait_asynccnt 0" ::: "memory");  // next tile landed
        __syncthreads();
        cur ^= 1;
    }

    // Epilogue. C/D layout: lane&15 -> N, VGPR v -> M = v + 8*(lane>=16).
    const int nlane = lane & 15;
    const int mAdd  = (lane >> 4) << 3;
#pragma unroll
    for (int i = 0; i < 2; ++i) {
#pragma unroll
        for (int j = 0; j < 4; ++j) {
            int   n_g = nBase + waveN * 64 + j * 16 + nlane;
            float bv  = bias[n_g];
#pragma unroll
            for (int v = 0; v < 8; ++v) {
                int   m_g = mBase + waveM * 32 + i * 16 + mAdd + v;
                float val = acc[i][j][v] + bv;
                if (mode == 0) {
                    int sec = n_g >> 10;          // 0=Q 1=K 2=V
                    int hc  = n_g & (HIDDEN - 1);
                    int h = hc >> 6, c = hc & 63;
                    int b = m_g >> 11, s = m_g & (SEQ - 1);
                    __bf16 ov = (__bf16)val;
                    size_t rowmaj = ((size_t)(b * NHEAD + h) * SEQ + s) * HSIZE + c;
                    if (sec == 0)      Qh[rowmaj] = ov;
                    else if (sec == 1) Kh[rowmaj] = ov;
                    else Vt[((size_t)(b * NHEAD + h) * HSIZE + c) * SEQ + s] = ov;
                } else {
                    Out[(size_t)m_g * HIDDEN + n_g] = val;
                }
            }
        }
    }
}

// ---------------------------------------------------------------------------
// Flash attention per (b,h). 128 threads = 4 waves, each wave owns 16 queries.
// Reference semantics: score[f(key), t(query)] kept where t <= f, replaced by
// -10000 otherwise; softmax over keys; ctx[t,c] = sum_f P[f,t] V[f,c].
// ---------------------------------------------------------------------------
__global__ __launch_bounds__(128) void attn_flash_wmma(
    const __bf16* __restrict__ Qh, const __bf16* __restrict__ Kh,
    const __bf16* __restrict__ Vt, __bf16* __restrict__ Ctx) {
    __shared__ __bf16 plds[4 * 16 * LDK];  // per-wave P transpose strip

    const int tid  = threadIdx.x;
    const int lane = tid & 31;
    const int w    = tid >> 5;
    const int bh   = blockIdx.y;
    const int q0   = blockIdx.x * 64 + w * 16;
    const int b = bh >> 4, h = bh & 15;

    const __bf16* Qb = Qh + (size_t)bh * SEQ * HSIZE;  // [s][c]
    const __bf16* Kb = Kh + (size_t)bh * SEQ * HSIZE;  // [s][c]
    const __bf16* Vb = Vt + (size_t)bh * HSIZE * SEQ;  // [c][s]

    // Q fragments: 16 queries x 64 dims -> two K=32 fragments
    v16bf qf0 = load_frag(Qb + (size_t)q0 * HSIZE, HSIZE, lane);
    v16bf qf1 = load_frag(Qb + (size_t)q0 * HSIZE + 32, HSIZE, lane);

    float rowMax[8], rowSum[8];
    v8f   o[4] = {};
#pragma unroll
    for (int v = 0; v < 8; ++v) { rowMax[v] = -3.0e38f; rowSum[v] = 0.0f; }

    const int nlane = lane & 15;
    const int mAdd  = (lane >> 4) << 3;
    __bf16* P = &plds[w * 16 * LDK];

    // keys allowed: f >= t  ->  start at the 32-aligned block containing q0
    for (int f0 = q0 & ~31; f0 < SEQ; f0 += 32) {
        if (f0 + 32 < SEQ) {  // hint next key/value block toward the caches
            __builtin_prefetch(Kb + (size_t)(f0 + 32 + nlane) * HSIZE, 0, 0);
            __builtin_prefetch(Vb + (size_t)nlane * SEQ + f0 + 32, 0, 0);
        }
        // scores S[16q x 32k] = Q x K^T : 4 WMMAs (2 key tiles x 2 c-halves)
        v16bf kf00 = load_frag(Kb + (size_t)f0 * HSIZE, HSIZE, lane);
        v16bf kf01 = load_frag(Kb + (size_t)f0 * HSIZE + 32, HSIZE, lane);
        v16bf kf10 = load_frag(Kb + (size_t)(f0 + 16) * HSIZE, HSIZE, lane);
        v16bf kf11 = load_frag(Kb + (size_t)(f0 + 16) * HSIZE + 32, HSIZE, lane);
        v8f s0 = {}, s1 = {};
        s0 = wmma_bf16(qf0, kf00, s0);
        s0 = wmma_bf16(qf1, kf01, s0);
        s1 = wmma_bf16(qf0, kf10, s1);
        s1 = wmma_bf16(qf1, kf11, s1);

        // online softmax (per query row m; rows live across 16 lanes)
        float alpha[8], p0[8], p1[8];
#pragma unroll
        for (int v = 0; v < 8; ++v) {
            int   t  = q0 + mAdd + v;
            float x0 = s0[v] * 0.125f;  // 1/sqrt(64)
            float x1 = s1[v] * 0.125f;
            if (t > f0 + nlane)      x0 = -10000.0f;  // MASKED_BIAS (replace)
            if (t > f0 + 16 + nlane) x1 = -10000.0f;
            float m = fmaxf(x0, x1);
#pragma unroll
            for (int d = 1; d < 16; d <<= 1) m = fmaxf(m, __shfl_xor(m, d, 32));
            float nm = fmaxf(rowMax[v], m);
            float al = __expf(rowMax[v] - nm);
            rowMax[v] = nm;
            alpha[v]  = al;
            x0 = __expf(x0 - nm);
            x1 = __expf(x1 - nm);
            p0[v] = x0; p1[v] = x1;
            float sv = x0 + x1;
#pragma unroll
            for (int d = 1; d < 16; d <<= 1) sv += __shfl_xor(sv, d, 32);
            rowSum[v] = rowSum[v] * al + sv;
        }
#pragma unroll
        for (int j = 0; j < 4; ++j)
#pragma unroll
            for (int v = 0; v < 8; ++v) o[j][v] *= alpha[v];

        // transpose P (C-layout -> A-layout) through per-wave LDS strip;
        // same-wave LDS ops are in-order, no barrier needed.
#pragma unroll
        for (int v = 0; v < 8; ++v) {
            P[(mAdd + v) * LDK + nlane]      = (__bf16)p0[v];
            P[(mAdd + v) * LDK + 16 + nlane] = (__bf16)p1[v];
        }
        v16bf pf = load_frag(P, LDK, lane);

        // ctx += P[16q x 32k] x V[32k x 64c] : 4 WMMAs, B frag straight from Vt
#pragma unroll
        for (int j = 0; j < 4; ++j) {
            v16bf vf = load_frag(Vb + (size_t)(j * 16) * SEQ + f0, SEQ, lane);
            o[j] = wmma_bf16(pf, vf, o[j]);
        }
    }

    // normalize and write ctx in [b, s, h*64+c] layout (bf16, GEMM2 input)
#pragma unroll
    for (int j = 0; j < 4; ++j) {
#pragma unroll
        for (int v = 0; v < 8; ++v) {
            int s   = q0 + mAdd + v;
            int col = h * HSIZE + j * 16 + nlane;
            Ctx[(size_t)(b * SEQ + s) * HIDDEN + col] = (__bf16)(o[j][v] / rowSum[v]);
        }
    }
}

// ---------------------------------------------------------------------------
extern "C" void kernel_launch(void* const* d_in, const int* in_sizes, int n_in,
                              void* d_out, int out_size, void* d_ws, size_t ws_size,
                              hipStream_t stream) {
    (void)in_sizes; (void)n_in; (void)out_size; (void)ws_size;
    const float* enc    = (const float*)d_in[0];  // [B,S,H] f32
    const float* w_attn = (const float*)d_in[1];  // [3H,H]  f32
    const float* b_attn = (const float*)d_in[2];  // [3H]    f32
    const float* w_proj = (const float*)d_in[3];  // [H,H]   f32
    const float* b_proj = (const float*)d_in[4];  // [H]     f32
    float* out = (float*)d_out;

    char*  ws  = (char*)d_ws;
    size_t off = 0;
    auto take = [&](size_t elems) -> __bf16* {
        __bf16* p = (__bf16*)(ws + off);
        off += ((elems * sizeof(__bf16)) + 255) & ~(size_t)255;
        return p;
    };
    __bf16* Xbf = take((size_t)TOK * HIDDEN);
    __bf16* Wa  = take((size_t)3 * HIDDEN * HIDDEN);
    __bf16* Wp  = take((size_t)HIDDEN * HIDDEN);
    __bf16* Qh  = take((size_t)TOK * HIDDEN);
    __bf16* Kh  = take((size_t)TOK * HIDDEN);
    __bf16* Vt  = take((size_t)TOK * HIDDEN);
    __bf16* Ctx = take((size_t)TOK * HIDDEN);

    cvt_f32_to_bf16<<<(TOK * HIDDEN) / 1024, 256, 0, stream>>>(enc, Xbf, TOK * HIDDEN);
    cvt_f32_to_bf16<<<(3 * HIDDEN * HIDDEN) / 1024, 256, 0, stream>>>(w_attn, Wa,
                                                                      3 * HIDDEN * HIDDEN);
    cvt_f32_to_bf16<<<(HIDDEN * HIDDEN) / 1024, 256, 0, stream>>>(w_proj, Wp,
                                                                  HIDDEN * HIDDEN);

    // QKV projection: [4096,1024] x [3072,1024]^T, scatter to Q/K/Vt layouts
    gemm_bf16_wmma<<<dim3(3 * HIDDEN / 128, TOK / 128), 256, 0, stream>>>(
        Xbf, Wa, b_attn, 3 * HIDDEN, 0, Qh, Kh, Vt, nullptr);

    // attention: one block per (64 queries, b*h)
    attn_flash_wmma<<<dim3(SEQ / 64, BATCH * NHEAD), 128, 0, stream>>>(Qh, Kh, Vt, Ctx);

    // output projection: [4096,1024] x [1024,1024]^T -> f32 out
    gemm_bf16_wmma<<<dim3(HIDDEN / 128, TOK / 128), 256, 0, stream>>>(
        Ctx, Wp, b_proj, HIDDEN, 1, nullptr, nullptr, nullptr, out);
}
